// ILSTM_SV_12077448036824
// MI455X (gfx1250) — matrix operand order repaired and checked by reading, not compiled, not run on hardware
//
#include <hip/hip_runtime.h>
#include <hip/hip_bf16.h>
#include <math.h>

// ---------------------------------------------------------------------------
// ILSTM_SV for MI455X (gfx1250, wave32).
// Dominant cost: 64 sequential steps of a fused [1024 x 1056] x [1056 x 4096]
// bf16 WMMA GEMM (i|f|o gates + block-diagonal j-transition folded in as a
// 4th N-block). Weights are packed once to bf16 N-major (~8.6 MB, L2-resident).
// GEMM is software-pipelined with double-buffered LDS; global->LDS staging
// uses the CDNA5 async-to-LDS DMA path (ASYNCcnt) when available.
// ---------------------------------------------------------------------------

typedef __attribute__((ext_vector_type(16))) __bf16 v16bf;
typedef __attribute__((ext_vector_type(8)))  float  v8f;
typedef __attribute__((ext_vector_type(4)))  int    v4i;

#define B_   1024
#define T_   64
#define D_   16
#define H_   64
#define DH   1024          // D*H
#define KTOT 1056          // 16 (x) + 1024 (h) padded to multiple of 32
#define NTOT 4096          // i | f | o | j
#define NK   (KTOT / 32)   // 33 k-iterations

#if __has_builtin(__builtin_amdgcn_global_load_async_to_lds_b128)
#define USE_ASYNC_LDS 1
#else
#define USE_ASYNC_LDS 0
#endif

// builtin expects: (int4 AS1*, int4 AS3*, imm offset, imm cpol)
#define GLD_SRC(p) ((__attribute__((address_space(1))) v4i*)(p))
#define GLD_DST(p) ((__attribute__((address_space(3))) v4i*)(p))

__device__ __forceinline__ void wait_async0() {
#if __has_builtin(__builtin_amdgcn_s_wait_asynccnt)
  __builtin_amdgcn_s_wait_asynccnt(0);
#else
  asm volatile("s_wait_asynccnt 0x0" ::: "memory");
#endif
}

__device__ __forceinline__ unsigned short f2bf(float f) {
  union { float f; unsigned u; } v; v.f = f;
  unsigned r = v.u + 0x7FFFu + ((v.u >> 16) & 1u);   // round-to-nearest-even
  return (unsigned short)(r >> 16);
}
__device__ __forceinline__ float sigmf(float x) { return 1.0f / (1.0f + __expf(-x)); }

// ---------------------------------------------------------------------------
// Pack Wall^T bf16 [NTOT][KTOT]: N-blocks = Wi | Wf | Wo | (W_j, U_j block-diag)
// ---------------------------------------------------------------------------
__global__ void pack_weights(const float* __restrict__ Wi, const float* __restrict__ Wf,
                             const float* __restrict__ Wo, const float* __restrict__ Wj,
                             const float* __restrict__ Uj, unsigned short* __restrict__ Wt) {
  long idx = (long)blockIdx.x * blockDim.x + threadIdx.x;
  if (idx >= (long)NTOT * KTOT) return;
  int n = (int)(idx / KTOT), k = (int)(idx % KTOT);
  float v = 0.f;
  if (k < 1040) {
    if (n < 3072) {
      const float* W = (n < 1024) ? Wi : (n < 2048 ? Wf : Wo);
      v = W[(size_t)k * 1024 + (n & 1023)];
    } else {
      int nj = n - 3072, dj = nj >> 6, hh = nj & 63;
      if (k < 16) { if (k == dj) v = Wj[dj * 64 + hh]; }             // x_t * W_j[d,0,:]
      else {
        int r = k - 16, dr = r >> 6, hr = r & 63;
        if (dr == dj) v = Uj[((size_t)dj * 64 + hr) * 64 + hh];      // h[:,d,:] @ U_j[d]
      }
    }
  }
  Wt[idx] = f2bf(v);
}

__global__ void pack_bias(const float* bi, const float* bf, const float* bo,
                          const float* bj, float* ball) {
  int n = blockIdx.x * blockDim.x + threadIdx.x;
  if (n >= NTOT) return;
  ball[n] = (n < 1024) ? bi[n] : (n < 2048) ? bf[n - 1024]
          : (n < 3072) ? bo[n - 2048] : bj[n - 3072];
}

__global__ void init_state(const float* __restrict__ x, unsigned short* __restrict__ gi,
                           float* __restrict__ c) {
  int idx = blockIdx.x * blockDim.x + threadIdx.x;
  if (idx < B_ * KTOT) {
    int b = idx / KTOT, col = idx % KTOT;
    unsigned short v = 0;
    if (col < 16) v = f2bf(x[(size_t)b * T_ * D_ + col]);  // x[:,0,:]; h0 = 0
    gi[idx] = v;
  }
  if (idx < B_ * DH) c[idx] = 0.f;
}

// ---------------------------------------------------------------------------
// Gate GEMM: preact[B_][NTOT] = gi[B_][KTOT](bf16) @ Wt^T   (Wt is N-major)
// 128x128 tile / block; 8 wave32s in 2(M)x4(N); each wave 4x2 WMMA tiles.
// LDS rows padded to 40 halves (20-bank stride -> conflict-free b128 reads).
// Double-buffered LDS, one barrier per k-iteration:
//   - issue DMA for tile kt+1 into buf[cur^1]
//   - ds_load fragments of tile kt from buf[cur], 8x WMMA (consumes reads
//     before the wave can signal the barrier -> no read/write race)
//   - s_wait_asynccnt 0, barrier
// ---------------------------------------------------------------------------
__global__ __launch_bounds__(256) void gemm_gates(
    const unsigned short* __restrict__ gi,   // [B_][KTOT] bf16
    const unsigned short* __restrict__ Wt,   // [NTOT][KTOT] bf16
    float* __restrict__ preact) {            // [B_][NTOT] f32
  __shared__ unsigned short sA[2][128 * 40];
  __shared__ unsigned short sB[2][128 * 40];
  const int tid  = threadIdx.x;
  const int lane = tid & 31;
  const int wave = tid >> 5;
  const int wm   = wave >> 2;        // 0..1  -> 64 M-rows
  const int wn   = wave & 3;         // 0..3  -> 32 N-cols
  const int m0   = blockIdx.y * 128;
  const int n0   = blockIdx.x * 128;
  const int half = lane >> 4;
  const int l16  = lane & 15;
  const int row  = tid >> 1;         // 0..127: tile row to stage
  const int chk  = tid & 1;          // 16-half chunk

  // per-thread staging addresses (one 16B chunk of A and of B per k-tile)
  const unsigned short* gA = gi + (size_t)(m0 + row) * KTOT + chk * 16;
  const unsigned short* gB = Wt + (size_t)(n0 + row) * KTOT + chk * 16;
  const int ldsoff = row * 40 + chk * 16;

  v8f acc[4][2] = {};

  // ---- prologue: stage tile 0 into buffer 0 ----
#if USE_ASYNC_LDS
  __builtin_amdgcn_global_load_async_to_lds_b128(GLD_SRC(gA), GLD_DST(&sA[0][ldsoff]), 0, 0);
  __builtin_amdgcn_global_load_async_to_lds_b128(GLD_SRC(gB), GLD_DST(&sB[0][ldsoff]), 0, 0);
  wait_async0();
#else
  *(uint4*)(&sA[0][ldsoff]) = *(const uint4*)gA;
  *(uint4*)(&sB[0][ldsoff]) = *(const uint4*)gB;
#endif
  __syncthreads();

  for (int kt = 0; kt < NK; ++kt) {
    const int cur = kt & 1;
    const bool pre = (kt + 1 < NK);

    // ---- issue staging of tile kt+1 into the other buffer ----
#if USE_ASYNC_LDS
    if (pre) {
      __builtin_amdgcn_global_load_async_to_lds_b128(
          GLD_SRC(gA + (size_t)(kt + 1) * 32), GLD_DST(&sA[cur ^ 1][ldsoff]), 0, 0);
      __builtin_amdgcn_global_load_async_to_lds_b128(
          GLD_SRC(gB + (size_t)(kt + 1) * 32), GLD_DST(&sB[cur ^ 1][ldsoff]), 0, 0);
    }
#else
    uint4 pfa, pfb;
    if (pre) {
      pfa = *(const uint4*)(gA + (size_t)(kt + 1) * 32);
      pfb = *(const uint4*)(gB + (size_t)(kt + 1) * 32);
    }
#endif

    // ---- fragments of tile kt ----
    const unsigned short* sAc = sA[cur];
    const unsigned short* sBc = sB[cur];
    v16bf afr[4], bfr[2];
    // A frag: lane row M=l16, element e -> K = (e&8)*2 + half*8 + (e&7)
#pragma unroll
    for (int mi = 0; mi < 4; ++mi) {
      const int M = wm * 64 + mi * 16 + l16;
      uint4* ap = (uint4*)&afr[mi];
      ap[0] = *(const uint4*)(sAc + M * 40 + half * 8);
      ap[1] = *(const uint4*)(sAc + M * 40 + 16 + half * 8);
    }
    // B frag: lane col N=l16, element e -> K = (e&8)*2 + half*8 + (e&7)
#pragma unroll
    for (int ni = 0; ni < 2; ++ni) {
      const int N = wn * 32 + ni * 16 + l16;
      uint4* bp = (uint4*)&bfr[ni];
      bp[0] = *(const uint4*)(sBc + N * 40 + half * 8);
      bp[1] = *(const uint4*)(sBc + N * 40 + 16 + half * 8);
    }
#pragma unroll
    for (int mi = 0; mi < 4; ++mi)
#pragma unroll
      for (int ni = 0; ni < 2; ++ni)
        acc[mi][ni] = __builtin_amdgcn_wmma_f32_16x16x32_bf16(
            false, afr[mi], false, bfr[ni], (short)0, acc[mi][ni], false, false);

    // ---- complete staging of tile kt+1, make it visible ----
#if USE_ASYNC_LDS
    if (pre) wait_async0();
#else
    if (pre) {
      *(uint4*)(&sA[cur ^ 1][ldsoff]) = pfa;
      *(uint4*)(&sB[cur ^ 1][ldsoff]) = pfb;
    }
#endif
    __syncthreads();
  }

  // C/D layout: VGPR r -> (M = r + 8*half, N = l16)
#pragma unroll
  for (int mi = 0; mi < 4; ++mi)
#pragma unroll
    for (int ni = 0; ni < 2; ++ni) {
      const int rbase = m0 + wm * 64 + mi * 16 + 8 * half;
      const int ccol  = n0 + wn * 32 + ni * 16 + l16;
#pragma unroll
      for (int r = 0; r < 8; ++r)
        preact[(size_t)(rbase + r) * NTOT + ccol] = acc[mi][ni][r];
    }
}

// ---------------------------------------------------------------------------
// Per-step elementwise: gates -> c,h update; emit outputs[:,t] and next gi.
// ---------------------------------------------------------------------------
__global__ void update_step(const float* __restrict__ preact, const float* __restrict__ ball,
                            float* __restrict__ c, const float* __restrict__ x,
                            float* __restrict__ outputs, unsigned short* __restrict__ gi,
                            int t) {
  int idx = blockIdx.x * blockDim.x + threadIdx.x;
  if (idx >= B_ * DH) return;
  int b = idx >> 10, k = idx & 1023;
  const float* pb = preact + (size_t)b * NTOT;
  float iv = sigmf(pb[k]        + ball[k]);
  float fv = sigmf(pb[1024 + k] + ball[1024 + k]);
  float ov = sigmf(pb[2048 + k] + ball[2048 + k]);
  float jv = tanhf(pb[3072 + k] + ball[3072 + k]);
  float cv = c[idx] * fv + iv * jv;
  c[idx] = cv;
  float hv = ov * tanhf(cv);
  outputs[((size_t)b * T_ + t) * DH + k] = hv;
  gi[(size_t)b * KTOT + 16 + k] = f2bf(hv);
  if (k < 16 && t + 1 < T_)
    gi[(size_t)b * KTOT + k] = f2bf(x[(size_t)b * T_ * D_ + (t + 1) * D_ + k]);
}

// ---------------------------------------------------------------------------
// Attention tails: one block per batch element (memory-bound, trivial cost).
// ---------------------------------------------------------------------------
__global__ __launch_bounds__(256) void tail_kernel(
    const float* __restrict__ outputs, const float* __restrict__ x,
    const float* __restrict__ Wfv_w, const float* __restrict__ Wfv_b,
    const float* __restrict__ Wpo_w, const float* __restrict__ Wpo_b,
    const float* __restrict__ Wpv_w, const float* __restrict__ Wpv_b,
    const float* __restrict__ Wap_w, const float* __restrict__ Wap_b,
    const float* __restrict__ Wto_w, const float* __restrict__ Wto_b,
    const float* __restrict__ Wtv_w, const float* __restrict__ Wtv_b,
    const float* __restrict__ Wat_w, const float* __restrict__ Wat_b,
    const float* __restrict__ Wfin_w, const float* __restrict__ Wfin_b,
    float* __restrict__ out) {
  __shared__ float s_at[1024];     // mulit_FV_aten[t][d]
  __shared__ float s_buf[1024];    // fv / pred_out (reused)
  __shared__ float s_tao[4096];    // tao[t][h]
  __shared__ float s_fc[16], s_ap[16], s_tfc[64], s_tat[64], s_inv[16], s_pp;

  const int b = blockIdx.x, tid = threadIdx.x;
  const float* ob = outputs + (size_t)b * T_ * DH;

  // fv = outputs @ Wfv + b  -> exp
  for (int i = tid; i < 1024; i += 256) {
    int t = i >> 4, d = i & 15;
    const float* p = ob + (size_t)t * DH + d * 64;
    float a = Wfv_b[0];
#pragma unroll
    for (int h = 0; h < 64; ++h) a += p[h] * Wfv_w[h];
    s_buf[i] = __expf(a);
  }
  __syncthreads();
  if (tid < 16) {                       // normalize over T
    float s = 0.f;
    for (int t = 0; t < 64; ++t) s += s_buf[t * 16 + tid];
    s_inv[tid] = 1.f / s;
  }
  __syncthreads();
  float* mfv_out = out + 1024 + (size_t)b * 1024;
  for (int i = tid; i < 1024; i += 256) {
    float a = s_buf[i] * s_inv[i & 15];
    s_at[i] = a;
    mfv_out[i] = a;                     // mulit_FV_aten [B,T,D]
  }
  __syncthreads();
  // predictor branch: pred_out[d][h] = sum_t mfi * Wpo_w[t] + Wpo_b
  for (int i = tid; i < 1024; i += 256) {
    int d = i >> 6;
    float a = 0.f;
    for (int t = 0; t < 64; ++t) a += s_at[t * 16 + d] * ob[(size_t)t * DH + i] * Wpo_w[t];
    s_buf[i] = a + Wpo_b[0];
  }
  __syncthreads();
  if (tid < 16) {
    int d = tid;
    float fc = Wpv_b[0], ap = Wap_b[0];
    for (int h = 0; h < 64; ++h) {
      float pv = s_buf[d * 64 + h];
      float ht = ob[(size_t)63 * DH + d * 64 + h];   // h_T
      fc += pv * Wpv_w[h] + ht * Wpv_w[64 + h];
      ap += pv * Wap_w[h] + ht * Wap_w[64 + h];
    }
    s_fc[d] = fc; s_ap[d] = __expf(ap);
  }
  __syncthreads();
  if (tid == 0) {
    float s = 0.f;
    for (int d = 0; d < 16; ++d) s += s_ap[d];
    float inv = 1.f / s, pp = 0.f;
    for (int d = 0; d < 16; ++d) {
      float a = s_ap[d] * inv;
      out[1024 + 1048576 + (size_t)b * 16 + d] = a;  // predicor_aten
      pp += a * s_fc[d];
    }
    s_pp = pp;
  }
  // temporal branch: tao[t][h] = sum_d mfi * Wto_w[d] + Wto_b
  for (int i = tid; i < 4096; i += 256) {
    int t = i >> 6, h = i & 63;
    float a = 0.f;
#pragma unroll
    for (int d = 0; d < 16; ++d) a += s_at[t * 16 + d] * ob[(size_t)t * DH + d * 64 + h] * Wto_w[d];
    s_tao[i] = a + Wto_b[0];
  }
  __syncthreads();
  if (tid < 64) {
    int t = tid;
    float fc = Wtv_b[0], at = Wat_b[0];
    for (int h = 0; h < 64; ++h) { float v = s_tao[t * 64 + h]; fc += v * Wtv_w[h]; at += v * Wat_w[h]; }
    for (int d = 0; d < 16; ++d) {
      float xv = x[(size_t)b * T_ * D_ + t * 16 + d];
      fc += xv * Wtv_w[64 + d]; at += xv * Wat_w[64 + d];
    }
    s_tfc[t] = fc; s_tat[t] = __expf(at);
  }
  __syncthreads();
  if (tid == 0) {
    float s = 0.f;
    for (int t = 0; t < 64; ++t) s += s_tat[t];
    float inv = 1.f / s, tp = 0.f;
    for (int t = 0; t < 64; ++t) {
      float a = s_tat[t] * inv;
      out[1024 + 1048576 + 16384 + (size_t)b * 64 + t] = a;  // temporal_aten
      tp += a * s_tfc[t];
    }
    out[b] = s_pp * Wfin_w[0] + tp * Wfin_w[1] + Wfin_b[0];  // prediction
  }
}

// ---------------------------------------------------------------------------
extern "C" void kernel_launch(void* const* d_in, const int* in_sizes, int n_in,
                              void* d_out, int out_size, void* d_ws, size_t ws_size,
                              hipStream_t stream) {
  (void)in_sizes; (void)n_in; (void)out_size; (void)ws_size;
  const float* x     = (const float*)d_in[0];
  const float* W_j   = (const float*)d_in[1];
  const float* U_j   = (const float*)d_in[2];
  const float* b_j   = (const float*)d_in[3];
  const float* Wi_w  = (const float*)d_in[4];
  const float* Wi_b  = (const float*)d_in[5];
  const float* Wf_w  = (const float*)d_in[6];
  const float* Wf_b  = (const float*)d_in[7];
  const float* Wo_w  = (const float*)d_in[8];
  const float* Wo_b  = (const float*)d_in[9];
  const float* Wfv_w = (const float*)d_in[10];
  const float* Wfv_b = (const float*)d_in[11];
  const float* Wpo_w = (const float*)d_in[12];
  const float* Wpo_b = (const float*)d_in[13];
  const float* Wpv_w = (const float*)d_in[14];
  const float* Wpv_b = (const float*)d_in[15];
  const float* Wap_w = (const float*)d_in[16];
  const float* Wap_b = (const float*)d_in[17];
  const float* Wto_w = (const float*)d_in[18];
  const float* Wto_b = (const float*)d_in[19];
  const float* Wtv_w = (const float*)d_in[20];
  const float* Wtv_b = (const float*)d_in[21];
  const float* Wat_w = (const float*)d_in[22];
  const float* Wat_b = (const float*)d_in[23];
  const float* Wfin_w = (const float*)d_in[24];
  const float* Wfin_b = (const float*)d_in[25];
  float* out = (float*)d_out;

  // Workspace carve-up (~287 MB total)
  char* ws = (char*)d_ws;
  unsigned short* Wt   = (unsigned short*)ws; ws += (size_t)NTOT * KTOT * 2;  // 8.65 MB
  unsigned short* gi   = (unsigned short*)ws; ws += (size_t)B_ * KTOT * 2;    // 2.16 MB
  float* ball          = (float*)ws;          ws += (size_t)NTOT * 4;         // 16 KB
  float* preact        = (float*)ws;          ws += (size_t)B_ * NTOT * 4;    // 16 MB
  float* cbuf          = (float*)ws;          ws += (size_t)B_ * DH * 4;      // 4 MB
  float* outputs       = (float*)ws;          ws += (size_t)B_ * T_ * DH * 4; // 256 MB

  pack_weights<<<((long)NTOT * KTOT + 255) / 256, 256, 0, stream>>>(Wi_w, Wf_w, Wo_w, W_j, U_j, Wt);
  pack_bias<<<NTOT / 256, 256, 0, stream>>>(Wi_b, Wf_b, Wo_b, b_j, ball);
  init_state<<<(B_ * KTOT + 255) / 256, 256, 0, stream>>>(x, gi, cbuf);

  dim3 ggrid(NTOT / 128, B_ / 128);  // 32 x 8 = 256 workgroups
  for (int t = 0; t < T_; ++t) {
    gemm_gates<<<ggrid, 256, 0, stream>>>(gi, Wt, preact);
    update_step<<<(B_ * DH) / 256, 256, 0, stream>>>(preact, ball, cbuf, x, outputs, gi, t);
  }

  tail_kernel<<<B_, 256, 0, stream>>>(outputs, x,
      Wfv_w, Wfv_b, Wpo_w, Wpo_b, Wpv_w, Wpv_b, Wap_w, Wap_b,
      Wto_w, Wto_b, Wtv_w, Wtv_b, Wat_w, Wat_b, Wfin_w, Wfin_b, out);
}
